// Transformer_24386824307062
// MI455X (gfx1250) — matrix operand order repaired and checked by reading, not compiled, not run on hardware
//
#include <hip/hip_runtime.h>

typedef _Float16 half_t;
typedef __attribute__((ext_vector_type(16))) _Float16 v16h;
typedef __attribute__((ext_vector_type(8)))  float    v8f;

#define B_  64
#define S_  512
#define D_  256
#define HQ_ 512
#define M_  (B_*S_)

#define GF_RELU   1
#define GF_F16OUT 2

union FragU { uint4 u[2]; v16h h; };

__device__ __forceinline__ v16h load_frag(const half_t* p0, const half_t* p1) {
  FragU f;
  f.u[0] = *(const uint4*)p0;
  f.u[1] = *(const uint4*)p1;
  return f.h;
}

__device__ __forceinline__ v8f wmma_f16(v16h a, v16h b, v8f c) {
  // D = A(16x32 f16) * B(32x16 f16) + C(16x16 f32)
  return __builtin_amdgcn_wmma_f32_16x16x32_f16(
      /*neg_a=*/false, a, /*neg_b=*/false, b,
      /*c_mod=*/(short)0, c, /*reuse_a=*/false, /*reuse_b=*/false);
}

// Async global->LDS copy of 16 bytes (tracked by ASYNCcnt, not LOADcnt).
__device__ __forceinline__ void async_copy_b128(const half_t* gaddr, half_t* lds) {
  unsigned l = (unsigned)(uintptr_t)lds;   // addrspace(3) offset
  asm volatile("global_load_async_to_lds_b128 %0, %1, off"
               :: "v"(l), "v"(gaddr) : "memory");
}

__device__ __forceinline__ void wait_asynccnt0() {
  asm volatile("s_wait_asynccnt 0x0" ::: "memory");
}

// -------------------------------------------------------------------------
// WMMA GEMM:  out[M,N] = A[M,K](f16) * W[N,K]^T(f16) + bias[N]
// block = 256 threads (8 waves); tile 128(M) x 128(N); K step 32.
// Double-buffered LDS filled with GLOBAL_LOAD_ASYNC_TO_LDS_B128; one
// barrier per k-step; copy of stage k+1 overlaps the 8 WMMAs of stage k.
// Each wave: 32x64 output (2x4 fragments of 16x16).
// -------------------------------------------------------------------------
__global__ void __launch_bounds__(256) gemm_wmma(
    const half_t* __restrict__ Abase, int lda, long long strideA,
    const half_t* __restrict__ Wbase, int ldw, long long strideW,
    const float*  __restrict__ biasBase, int strideBias,
    void* __restrict__ outBase, int ldo, long long strideOut,
    int M, int N, int K, int flags)
{
  __shared__ half_t As[2][128][40];   // +8 halves pad keeps 16B alignment
  __shared__ half_t Bs[2][128][40];

  const int tid    = threadIdx.x;
  const int lane   = tid & 31;
  const int wave   = tid >> 5;
  const int waveM  = wave >> 1;          // 0..3 -> 32 rows each
  const int waveN  = wave & 1;           // 0..1 -> 64 cols each
  const int half16 = (lane >> 4) & 1;
  const int l15    = lane & 15;

  const int bz = blockIdx.z;
  const int m0 = blockIdx.y * 128;
  const int n0 = blockIdx.x * 128;

  const half_t* A = Abase + (size_t)bz * (size_t)strideA;
  const half_t* W = Wbase + (size_t)bz * (size_t)strideW;
  const float* bias = biasBase ? biasBase + (size_t)bz * (size_t)strideBias : nullptr;

  // Pre-zero LDS rows for out-of-range M (async copies skip them via EXEC).
  if (m0 + 128 > M) {
    uint4 z = make_uint4(0u, 0u, 0u, 0u);
    #pragma unroll
    for (int st = 0; st < 2; ++st)
      #pragma unroll
      for (int i = 0; i < 2; ++i) {
        int chunk = tid + i * 256;
        int row   = chunk >> 2;
        int c8    = (chunk & 3) * 8;
        if (m0 + row >= M) *(uint4*)&As[st][row][c8] = z;
      }
  }

  auto issue_stage = [&](int st, int k0) {
    #pragma unroll
    for (int i = 0; i < 2; ++i) {            // A tile: 128 x 32 halves
      int chunk = tid + i * 256;
      int row   = chunk >> 2;
      int c8    = (chunk & 3) * 8;
      int grow  = m0 + row;
      if (grow < M)
        async_copy_b128(A + (size_t)grow * lda + k0 + c8, &As[st][row][c8]);
    }
    #pragma unroll
    for (int i = 0; i < 2; ++i) {            // B tile: 128 x 32 halves
      int chunk = tid + i * 256;
      int row   = chunk >> 2;
      int c8    = (chunk & 3) * 8;
      async_copy_b128(W + (size_t)(n0 + row) * ldw + k0 + c8, &Bs[st][row][c8]);
    }
  };

  v8f acc[2][4];
  v8f vzero = {};
  #pragma unroll
  for (int i = 0; i < 2; ++i)
    #pragma unroll
    for (int j = 0; j < 4; ++j) acc[i][j] = vzero;

  const int nk = K >> 5;
  issue_stage(0, 0);

  for (int kt = 0; kt < nk; ++kt) {
    wait_asynccnt0();       // our copies for stage kt have landed in LDS
    __syncthreads();        // everyone's landed; everyone done reading other buf
    if (kt + 1 < nk) issue_stage((kt + 1) & 1, (kt + 1) * 32);

    const int cur = kt & 1;
    v16h afr[2], bfr[4];
    #pragma unroll
    for (int sm = 0; sm < 2; ++sm) {
      int row = waveM * 32 + sm * 16 + l15;
      int kb  = half16 * 8;                 // A layout: K {0..7,16..23} / {8..15,24..31}
      afr[sm] = load_frag(&As[cur][row][kb], &As[cur][row][kb + 16]);
    }
    #pragma unroll
    for (int sn = 0; sn < 4; ++sn) {
      int col = waveN * 64 + sn * 16 + l15;
      int kb2 = half16 * 8;                 // B layout: K {0..7,16..23} / {8..15,24..31}
      bfr[sn] = load_frag(&Bs[cur][col][kb2], &Bs[cur][col][kb2 + 16]);
    }
    #pragma unroll
    for (int sm = 0; sm < 2; ++sm)
      #pragma unroll
      for (int sn = 0; sn < 4; ++sn)
        acc[sm][sn] = wmma_f16(afr[sm], bfr[sn], acc[sm][sn]);
  }

  // ---- epilogue: bias (+relu), store f32 or f16 ----
  #pragma unroll
  for (int sm = 0; sm < 2; ++sm) {
    #pragma unroll
    for (int sn = 0; sn < 4; ++sn) {
      int gcol = n0 + waveN * 64 + sn * 16 + l15;
      float bv = bias ? bias[gcol] : 0.f;
      #pragma unroll
      for (int r = 0; r < 8; ++r) {
        int grow = m0 + waveM * 32 + sm * 16 + r + half16 * 8;
        if (grow >= M) continue;
        float vv = acc[sm][sn][r] + bv;
        if (flags & GF_RELU) vv = fmaxf(vv, 0.f);
        size_t idx = (size_t)bz * (size_t)strideOut + (size_t)grow * ldo + gcol;
        if (flags & GF_F16OUT) ((half_t*)outBase)[idx] = (half_t)vv;
        else                   ((float*) outBase)[idx] = vv;
      }
    }
  }
}

// -------------------------------------------------------------------------
// Flash attention (non-causal): per (block of 128 q-rows, head, batch).
// Each wave: 16 query rows, streams 32-wide key tiles, online softmax.
// -------------------------------------------------------------------------
__global__ void __launch_bounds__(256) attention_flash(
    const half_t* __restrict__ q, const half_t* __restrict__ k,
    const half_t* __restrict__ v, half_t* __restrict__ attn)
{
  __shared__ half_t Vt[64][40];        // transposed V tile [n][k]
  __shared__ half_t Pw[8][16][40];     // per-wave probability tile

  const int tid    = threadIdx.x;
  const int lane   = tid & 31;
  const int w      = tid >> 5;
  const int half16 = (lane >> 4) & 1;
  const int l15    = lane & 15;

  const int b     = blockIdx.z;
  const int h     = blockIdx.y;
  const int qrow0 = blockIdx.x * 128 + w * 16;

  // Q fragments (16 rows x 64 K), K contiguous in memory
  v16h qfr[2];
  {
    int row = qrow0 + l15;
    const half_t* qp = q + ((size_t)(b * S_ + row) * HQ_) + h * 64;
    #pragma unroll
    for (int j = 0; j < 2; ++j) {
      int kb = j * 32 + half16 * 8;
      qfr[j] = load_frag(qp + kb, qp + kb + 16);
    }
  }

  v8f vzero = {};
  v8f oacc[4] = { vzero, vzero, vzero, vzero };
  float mrow[8], lrow[8];
  #pragma unroll
  for (int r = 0; r < 8; ++r) { mrow[r] = -1e30f; lrow[r] = 0.f; }

  const float scale = 0.125f;  // 1/sqrt(64)

  for (int kt = 0; kt < 16; ++kt) {
    const int s0 = kt * 32;
    __syncthreads();
    // stage V tile transposed: Vt[n][k] = V[s0+k][n]
    #pragma unroll
    for (int j = 0; j < 8; ++j) {
      int e  = j * 256 + tid;
      int sl = e >> 6;
      int n  = e & 63;
      Vt[n][sl] = v[((size_t)(b * S_ + s0 + sl) * HQ_) + h * 64 + n];
    }
    __syncthreads();

    // scores tile 16x32 = two 16x16 WMMA accumulators
    v8f sc[2] = { vzero, vzero };
    #pragma unroll
    for (int sub = 0; sub < 2; ++sub) {
      int col = s0 + sub * 16 + l15;
      const half_t* kp = k + ((size_t)(b * S_ + col) * HQ_) + h * 64;
      #pragma unroll
      for (int j = 0; j < 2; ++j) {
        int kb = j * 32 + half16 * 8;
        v16h kf = load_frag(kp + kb, kp + kb + 16);
        sc[sub] = wmma_f16(qfr[j], kf, sc[sub]);
      }
    }

    // online softmax update (row reductions within 16-lane halves)
    #pragma unroll
    for (int r = 0; r < 8; ++r) {
      float a0 = sc[0][r] * scale;
      float a1 = sc[1][r] * scale;
      float mt = fmaxf(a0, a1);
      #pragma unroll
      for (int off = 1; off < 16; off <<= 1)
        mt = fmaxf(mt, __shfl_xor(mt, off, 32));
      float nm    = fmaxf(mrow[r], mt);
      float alpha = __expf(mrow[r] - nm);
      float p0 = __expf(a0 - nm);
      float p1 = __expf(a1 - nm);
      float ps = p0 + p1;
      #pragma unroll
      for (int off = 1; off < 16; off <<= 1)
        ps += __shfl_xor(ps, off, 32);
      lrow[r] = lrow[r] * alpha + ps;
      mrow[r] = nm;
      #pragma unroll
      for (int sub = 0; sub < 4; ++sub) oacc[sub][r] *= alpha;
      int prow = r + half16 * 8;
      Pw[w][prow][l15]      = (half_t)p0;
      Pw[w][prow][16 + l15] = (half_t)p1;
    }

    // O += P(16x32) @ V(32x64)
    {
      int row = l15;
      int kb  = half16 * 8;
      v16h pf = load_frag(&Pw[w][row][kb], &Pw[w][row][kb + 16]);
      #pragma unroll
      for (int sub = 0; sub < 4; ++sub) {
        int col = sub * 16 + l15;
        int kb2 = half16 * 8;
        v16h vf = load_frag(&Vt[col][kb2], &Vt[col][kb2 + 16]);
        oacc[sub] = wmma_f16(pf, vf, oacc[sub]);
      }
    }
  }

  // normalize and store f16 attention output [B*S, 512]
  #pragma unroll
  for (int sub = 0; sub < 4; ++sub) {
    int col = h * 64 + sub * 16 + l15;
    #pragma unroll
    for (int r = 0; r < 8; ++r) {
      int grow = qrow0 + r + half16 * 8;
      attn[((size_t)(b * S_ + grow)) * HQ_ + col] = (half_t)(oacc[sub][r] / lrow[r]);
    }
  }
}

// -------------------------------------------------------------------------
// Fused residual add + LayerNorm over D=256; one wave per row.
// -------------------------------------------------------------------------
__global__ void __launch_bounds__(256) add_ln(
    const float* __restrict__ xin, const float* __restrict__ res,
    const float* __restrict__ g, const float* __restrict__ bb,
    float* __restrict__ enc, half_t* __restrict__ enc16)
{
  const int lane = threadIdx.x & 31;
  const int w    = threadIdx.x >> 5;
  const int row  = blockIdx.x * 8 + w;
  const size_t base = (size_t)row * D_;

  float vv[8];
  float s = 0.f;
  #pragma unroll
  for (int i = 0; i < 8; ++i) {
    int d = lane + i * 32;
    vv[i] = xin[base + d] + res[base + d];
    s += vv[i];
  }
  #pragma unroll
  for (int off = 1; off < 32; off <<= 1) s += __shfl_xor(s, off, 32);
  float mean = s * (1.f / 256.f);
  float vs = 0.f;
  #pragma unroll
  for (int i = 0; i < 8; ++i) { float dlt = vv[i] - mean; vs += dlt * dlt; }
  #pragma unroll
  for (int off = 1; off < 32; off <<= 1) vs += __shfl_xor(vs, off, 32);
  float rstd = rsqrtf(vs * (1.f / 256.f) + 1e-5f);
  #pragma unroll
  for (int i = 0; i < 8; ++i) {
    int d = lane + i * 32;
    float y = (vv[i] - mean) * rstd * g[d] + bb[d];
    enc[base + d]   = y;
    enc16[base + d] = (half_t)y;
  }
}

__global__ void f32_to_f16(const float* __restrict__ in, half_t* __restrict__ out, int n) {
  int i = blockIdx.x * 256 + threadIdx.x;
  if (i < n) out[i] = (half_t)in[i];
}

__global__ void init_enc(const float* __restrict__ x, float* __restrict__ enc,
                         half_t* __restrict__ enc16, int n) {
  int i = blockIdx.x * 256 + threadIdx.x;
  if (i < n) { float v = x[i]; enc[i] = v; enc16[i] = (half_t)v; }
}

// top-5 over sequence per (b, d) channel; output [b, d*5 + j] as f16
__global__ void __launch_bounds__(256) topk5(const float* __restrict__ enc,
                                             half_t* __restrict__ top16) {
  int i = blockIdx.x * 256 + threadIdx.x;   // 0..16383
  int b = i >> 8;
  int d = i & 255;
  float t0 = -1e30f, t1 = -1e30f, t2 = -1e30f, t3 = -1e30f, t4 = -1e30f;
  const float* pcol = enc + (size_t)b * S_ * D_ + d;
  for (int s = 0; s < S_; ++s) {
    float v = pcol[(size_t)s * D_];
    if (v > t4) {
      if (v > t0)      { t4 = t3; t3 = t2; t2 = t1; t1 = t0; t0 = v; }
      else if (v > t1) { t4 = t3; t3 = t2; t2 = t1; t1 = v; }
      else if (v > t2) { t4 = t3; t3 = t2; t2 = v; }
      else if (v > t3) { t4 = t3; t3 = v; }
      else             { t4 = v; }
    }
  }
  half_t* o = top16 + (size_t)b * 1280 + d * 5;
  o[0] = (half_t)t0; o[1] = (half_t)t1; o[2] = (half_t)t2;
  o[3] = (half_t)t3; o[4] = (half_t)t4;
}

// logits[b,k] = sigmoid(dot(hh[k][b], Wh2[k]) + bh2[k]); one wave per (k,b)
__global__ void __launch_bounds__(256) head_logits(
    const float* __restrict__ hh, const float* __restrict__ Wh2,
    const float* __restrict__ bh2, float* __restrict__ outp)
{
  int g    = blockIdx.x * 8 + (threadIdx.x >> 5);  // 0..767
  int lane = threadIdx.x & 31;
  int kk   = g >> 6;    // head 0..11
  int bb   = g & 63;    // batch
  const float* hrow = hh + ((size_t)(kk * 64 + bb)) * 256;
  const float* wrow = Wh2 + kk * 256;
  float s = 0.f;
  #pragma unroll
  for (int i = 0; i < 8; ++i) { int d = lane + i * 32; s += hrow[d] * wrow[d]; }
  #pragma unroll
  for (int off = 1; off < 32; off <<= 1) s += __shfl_xor(s, off, 32);
  if (lane == 0) {
    float z = s + bh2[kk];
    outp[bb * 12 + kk] = 1.f / (1.f + __expf(-z));
  }
}

// -------------------------------------------------------------------------
extern "C" void kernel_launch(void* const* d_in, const int* in_sizes, int n_in,
                              void* d_out, int out_size, void* d_ws, size_t ws_size,
                              hipStream_t stream) {
  (void)in_sizes; (void)n_in; (void)out_size; (void)ws_size;
  const float* x   = (const float*)d_in[0];
  const float* Wq  = (const float*)d_in[1];
  const float* bq  = (const float*)d_in[2];
  const float* Wk  = (const float*)d_in[3];
  const float* bk  = (const float*)d_in[4];
  const float* Wv  = (const float*)d_in[5];
  const float* bv  = (const float*)d_in[6];
  const float* Wo  = (const float*)d_in[7];
  const float* bo  = (const float*)d_in[8];
  const float* Wc1 = (const float*)d_in[9];
  const float* bc1 = (const float*)d_in[10];
  const float* Wc2 = (const float*)d_in[11];
  const float* bc2 = (const float*)d_in[12];
  const float* g1  = (const float*)d_in[13];
  const float* b1  = (const float*)d_in[14];
  const float* g2  = (const float*)d_in[15];
  const float* b2  = (const float*)d_in[16];
  const float* Wm  = (const float*)d_in[17];
  const float* bm  = (const float*)d_in[18];
  const float* Wh1 = (const float*)d_in[19];
  const float* bh1 = (const float*)d_in[20];
  const float* Wh2 = (const float*)d_in[21];
  const float* bh2 = (const float*)d_in[22];

  // ---- workspace partition ----
  char* p = (char*)d_ws;
  auto alloc = [&](size_t bytes) -> void* {
    void* r = (void*)p;
    p += (bytes + 255) & ~(size_t)255;
    return r;
  };
  half_t* Wq16  = (half_t*)alloc((size_t)4 * 512 * 256 * 2);
  half_t* Wk16  = (half_t*)alloc((size_t)4 * 512 * 256 * 2);
  half_t* Wv16  = (half_t*)alloc((size_t)4 * 512 * 256 * 2);
  half_t* Wo16  = (half_t*)alloc((size_t)4 * 256 * 512 * 2);
  half_t* Wc116 = (half_t*)alloc((size_t)4 * 256 * 256 * 2);
  half_t* Wc216 = (half_t*)alloc((size_t)4 * 256 * 256 * 2);
  half_t* Wm16  = (half_t*)alloc((size_t)256 * 1280 * 2);
  half_t* Wh116 = (half_t*)alloc((size_t)12 * 256 * 256 * 2);
  float*  enc    = (float*) alloc((size_t)M_ * 256 * 4);
  half_t* enc16  = (half_t*)alloc((size_t)M_ * 256 * 2);
  half_t* qb     = (half_t*)alloc((size_t)M_ * 512 * 2);
  half_t* kb2buf = (half_t*)alloc((size_t)M_ * 512 * 2);
  half_t* vb     = (half_t*)alloc((size_t)M_ * 512 * 2);
  half_t* attn16 = (half_t*)alloc((size_t)M_ * 512 * 2);
  float*  proj   = (float*) alloc((size_t)M_ * 256 * 4);
  half_t* top16  = (half_t*)alloc((size_t)64 * 1280 * 2);
  float*  outv   = (float*) alloc((size_t)64 * 256 * 4);
  half_t* out16  = (half_t*)alloc((size_t)64 * 256 * 2);
  float*  hh     = (float*) alloc((size_t)12 * 64 * 256 * 4);

  dim3 blk(256);
  auto conv = [&](const float* src, half_t* dst, int n) {
    f32_to_f16<<<dim3((n + 255) / 256), blk, 0, stream>>>(src, dst, n);
  };

  // ---- weight conversion to f16 ----
  conv(Wq,  Wq16,  4 * 512 * 256);
  conv(Wk,  Wk16,  4 * 512 * 256);
  conv(Wv,  Wv16,  4 * 512 * 256);
  conv(Wo,  Wo16,  4 * 256 * 512);
  conv(Wc1, Wc116, 4 * 256 * 256);
  conv(Wc2, Wc216, 4 * 256 * 256);
  conv(Wm,  Wm16,  256 * 1280);
  conv(Wh1, Wh116, 12 * 256 * 256);

  init_enc<<<dim3((M_ * 256 + 255) / 256), blk, 0, stream>>>(x, enc, enc16, M_ * 256);

  for (int i = 0; i < 4; ++i) {
    // QKV projections -> f16 [M, 512]
    gemm_wmma<<<dim3(512 / 128, M_ / 128, 1), blk, 0, stream>>>(
        enc16, 256, 0, Wq16 + (size_t)i * 512 * 256, 256, 0, bq + i * 512, 0,
        qb, 512, 0, M_, 512, 256, GF_F16OUT);
    gemm_wmma<<<dim3(512 / 128, M_ / 128, 1), blk, 0, stream>>>(
        enc16, 256, 0, Wk16 + (size_t)i * 512 * 256, 256, 0, bk + i * 512, 0,
        kb2buf, 512, 0, M_, 512, 256, GF_F16OUT);
    gemm_wmma<<<dim3(512 / 128, M_ / 128, 1), blk, 0, stream>>>(
        enc16, 256, 0, Wv16 + (size_t)i * 512 * 256, 256, 0, bv + i * 512, 0,
        vb, 512, 0, M_, 512, 256, GF_F16OUT);

    attention_flash<<<dim3(S_ / 128, 8, B_), blk, 0, stream>>>(qb, kb2buf, vb, attn16);

    // output projection -> f32 [M, 256]
    gemm_wmma<<<dim3(256 / 128, M_ / 128, 1), blk, 0, stream>>>(
        attn16, 512, 0, Wo16 + (size_t)i * 256 * 512, 512, 0, bo + i * 256, 0,
        proj, 256, 0, M_, 256, 512, 0);

    add_ln<<<dim3(M_ / 8), blk, 0, stream>>>(proj, enc, g1 + i * 256, b1 + i * 256, enc, enc16);

    // FFN (reuse qb as f16 hidden, kb2buf as f32 output)
    gemm_wmma<<<dim3(256 / 128, M_ / 128, 1), blk, 0, stream>>>(
        enc16, 256, 0, Wc116 + (size_t)i * 256 * 256, 256, 0, bc1 + i * 256, 0,
        qb, 256, 0, M_, 256, 256, GF_RELU | GF_F16OUT);
    gemm_wmma<<<dim3(256 / 128, M_ / 128, 1), blk, 0, stream>>>(
        qb, 256, 0, Wc216 + (size_t)i * 256 * 256, 256, 0, bc2 + i * 256, 0,
        (float*)kb2buf, 256, 0, M_, 256, 256, 0);

    add_ln<<<dim3(M_ / 8), blk, 0, stream>>>((float*)kb2buf, enc, g2 + i * 256, b2 + i * 256, enc, enc16);
  }

  // ---- tail ----
  topk5<<<dim3(64), blk, 0, stream>>>(enc, top16);

  // out = top @ Wm^T + bm : [64, 1280] x [256, 1280]^T
  gemm_wmma<<<dim3(256 / 128, 1, 1), blk, 0, stream>>>(
      top16, 1280, 0, Wm16, 1280, 0, bm, 0,
      outv, 256, 0, 64, 256, 1280, 0);
  conv(outv, out16, 64 * 256);

  // 12 batched heads: hh[k] = relu(out @ Wh1[k]^T + bh1[k]) : [64,256]
  gemm_wmma<<<dim3(256 / 128, 1, 12), blk, 0, stream>>>(
      out16, 256, 0, Wh116, 256, (long long)256 * 256, bh1, 256,
      hh, 256, (long long)64 * 256, 64, 256, 256, GF_RELU);

  head_logits<<<dim3(96), blk, 0, stream>>>(hh, Wh2, bh2, (float*)d_out);
}
